// Attention_27668179320934
// MI455X (gfx1250) — compile-verified, matrix operand order checked by reading, not run on hardware
//
#include <hip/hip_runtime.h>
#include <cmath>

typedef __attribute__((ext_vector_type(16))) _Float16 v16h;
typedef __attribute__((ext_vector_type(8)))  _Float16 v8h;
typedef __attribute__((ext_vector_type(4)))  _Float16 v4h;
typedef __attribute__((ext_vector_type(8)))  float    v8f;

#define ENC_DIM  1024
#define DEC_DIM  1024
#define FAN_IN   2048
#define BATCH    32
#define SRC_LEN  2048
#define ROWS     (SRC_LEN * BATCH)          // 65536 flattened (s,b) rows
#define MBLOCK   128                        // rows per workgroup
#define MT_STRIDE 1032                      // f16 elems per row slot (16B aligned)
#define GRP_STRIDE 8328                     // 8*MT_STRIDE + 72: lane stride = 4 banks (mod 64)
#define A_LDS_BYTES (16 * GRP_STRIDE * 2)   // 266496 B
#define SMEM_BYTES  (A_LDS_BYTES + MBLOCK * 4)

// ---------------------------------------------------------------------------
// Kernel 1: convert the encoder half of attn_w (cols 1024..2047) to f16,
// row-major [d][k]. 2 MiB -> L2 resident for the whole main GEMM.
// ---------------------------------------------------------------------------
__global__ void wconv_kernel(const float* __restrict__ attn_w,
                             _Float16* __restrict__ wf16) {
    int gid = blockIdx.x * 256 + threadIdx.x;     // 262144 quads
    int d = gid >> 8;                             // 256 quads per row
    int k = (gid & 255) << 2;
    float4 f = *(const float4*)(attn_w + (size_t)d * FAN_IN + DEC_DIM + k);
    v4h hv = { (_Float16)f.x, (_Float16)f.y, (_Float16)f.z, (_Float16)f.w };
    *(v4h*)(wf16 + (size_t)d * ENC_DIM + k) = hv;
}

// ---------------------------------------------------------------------------
// Kernel 2: hpartT[d][b] = attn_b[d] + hidden[b,:] . attn_w[d, :1024]
// The hidden half of the concat is s-invariant: hoisted out of the big GEMM.
// ---------------------------------------------------------------------------
__global__ void hpart_kernel(const float* __restrict__ hidden,
                             const float* __restrict__ attn_w,
                             const float* __restrict__ attn_b,
                             float* __restrict__ hpartT) {
    int gid = blockIdx.x * 256 + threadIdx.x;     // 32768
    int d = gid >> 5, b = gid & 31;
    const float* wr = attn_w + (size_t)d * FAN_IN;
    const float* hr = hidden + (size_t)b * DEC_DIM;
    float s = attn_b[d];
    for (int f = 0; f < DEC_DIM; f += 4) {
        float4 wv = *(const float4*)(wr + f);
        float4 hv = *(const float4*)(hr + f);
        s += wv.x * hv.x + wv.y * hv.y + wv.z * hv.z + wv.w * hv.w;
    }
    hpartT[d * BATCH + b] = s;
}

// ---------------------------------------------------------------------------
// Kernel 3: fused GEMM (f16 WMMA, f32 accum) + tanh + v-dot reduction.
// logits[b][s] = sum_d v[d] * tanh(enc[s,b,:] . Wenc[d,:] + hpartT[d][b])
// LDS layout for A: [row%16][row/16][MT_STRIDE] f16, group stride GRP_STRIDE,
// so a lane's 8 m-tile fragments are reachable with immediate DS offsets.
// ---------------------------------------------------------------------------
__global__ void __launch_bounds__(256, 1)
attn_main_kernel(const float* __restrict__ enc,
                 const _Float16* __restrict__ wf16,
                 const float* __restrict__ hpartT,
                 const float* __restrict__ vvec,
                 float* __restrict__ logits) {
    extern __shared__ char smem[];
    _Float16* ldsA = (_Float16*)smem;
    float*    sLog = (float*)(smem + A_LDS_BYTES);

    const int tid = threadIdx.x;
    const int r0  = blockIdx.x * MBLOCK;

    if (tid < MBLOCK) sLog[tid] = 0.0f;

    // Stage 128x1024 f32 enc rows into LDS as f16 (coalesced float4 loads).
    for (int it = 0; it < (MBLOCK * (ENC_DIM / 4)) / 256; ++it) {
        int idx = tid + it * 256;
        int row = idx >> 8;                 // 256 quads per row
        int k   = (idx & 255) << 2;
        float4 f = *(const float4*)(enc + (size_t)(r0 + row) * ENC_DIM + k);
        v4h hv = { (_Float16)f.x, (_Float16)f.y, (_Float16)f.z, (_Float16)f.w };
        *(v4h*)(ldsA + (row & 15) * GRP_STRIDE + (row >> 4) * MT_STRIDE + k) = hv;
    }
    __syncthreads();

    const int wave = tid >> 5;
    const int lane = tid & 31;
    const int h    = lane >> 4;     // half-wave: K-phase / high rows
    const int nl   = lane & 15;     // M (A) / N (B) index within tile
    const int kb   = h * 8;         // K base per the 16-bit fragment layout

    // Lane-fixed LDS base: fragment (mt, k0) lives at + mt*MT_STRIDE + k0 (+16).
    const _Float16* Abase = ldsA + nl * GRP_STRIDE + kb;

    auto loadA = [&](int mt, int k0) -> v16h {
        const _Float16* p = Abase + mt * MT_STRIDE + k0;
        v8h lo = *(const v8h*)(p);
        v8h hi = *(const v8h*)(p + 16);
        return __builtin_shufflevector(lo, hi,
            0,1,2,3,4,5,6,7,8,9,10,11,12,13,14,15);
    };

    for (int j = 0; j < 8; ++j) {                 // 8 N-tiles per wave
        const int n0 = (wave * 8 + j) * 16;
        const int d  = n0 + nl;                   // this lane's output column
        const _Float16* Bp = wf16 + (size_t)d * ENC_DIM + kb;

        auto loadB = [&](int k0) -> v16h {
            v8h lo = *(const v8h*)(Bp + k0);
            v8h hi = *(const v8h*)(Bp + k0 + 16);
            return __builtin_shufflevector(lo, hi,
                0,1,2,3,4,5,6,7,8,9,10,11,12,13,14,15);
        };

        v8f acc[8] = {};                          // 8 M-tiles register-blocked

        // Software pipeline: B double-buffered per k-step, A rotated one
        // m-tile ahead so each WMMA consumes a fragment loaded one WMMA ago.
        v16h bf = loadB(0);
        v16h a_cur = loadA(0, 0);
        for (int k0 = 0; k0 < ENC_DIM; k0 += 32) {
            v16h bf_n = loadB((k0 + 32) & (ENC_DIM - 1));
            #pragma unroll
            for (int mt = 0; mt < 8; ++mt) {
                v16h a_next;
                if (mt < 7) a_next = loadA(mt + 1, k0);
                else        a_next = loadA(0, (k0 + 32) & (ENC_DIM - 1));
                acc[mt] = __builtin_amdgcn_wmma_f32_16x16x32_f16(
                    false, a_cur, false, bf, (short)0, acc[mt], false, false);
                a_cur = a_next;
            }
            bf = bf_n;
        }

        // Fused epilogue: +hpartT, tanh, *v[d]; reduce 16 cols per row via
        // xor-butterfly (xor 1,2,4,8 stays inside each 16-lane half), then
        // accumulate rows across waves with LDS float atomics.
        // Row of acc[mt] elem i = r0 + mt*16 + i + 8h -> b = i + 8h + 16*(mt&1).
        const float vc = vvec[d];
        const float* hpt = hpartT + (size_t)d * BATCH;
        float4 a0 = *(const float4*)(hpt + 8 * h);
        float4 a1 = *(const float4*)(hpt + 8 * h + 4);
        float4 b0 = *(const float4*)(hpt + 16 + 8 * h);
        float4 b1 = *(const float4*)(hpt + 16 + 8 * h + 4);
        float hpA[8] = { a0.x,a0.y,a0.z,a0.w, a1.x,a1.y,a1.z,a1.w };
        float hpB[8] = { b0.x,b0.y,b0.z,b0.w, b1.x,b1.y,b1.z,b1.w };
        #pragma unroll
        for (int mt = 0; mt < 8; ++mt) {
            #pragma unroll
            for (int i = 0; i < 8; ++i) {
                float hp = (mt & 1) ? hpB[i] : hpA[i];
                float s  = vc * tanhf(acc[mt][i] + hp);
                s += __shfl_xor(s, 1, 32);
                s += __shfl_xor(s, 2, 32);
                s += __shfl_xor(s, 4, 32);
                s += __shfl_xor(s, 8, 32);
                if (nl == 0) atomicAdd(&sLog[mt * 16 + i + 8 * h], s);
            }
        }
    }
    __syncthreads();

    if (tid < MBLOCK) {
        int r = r0 + tid;
        int s_idx = r >> 5;         // s
        int b     = r & 31;         // batch
        logits[(size_t)b * SRC_LEN + s_idx] = sLog[tid];
    }
}

// ---------------------------------------------------------------------------
// Kernel 4: row softmax over s for each b.
// ---------------------------------------------------------------------------
__global__ void softmax_kernel(const float* __restrict__ logits,
                               float* __restrict__ out) {
    __shared__ float red[256];
    const int b = blockIdx.x, tid = threadIdx.x;
    const float* row = logits + (size_t)b * SRC_LEN;

    float mx = -1e30f;
    for (int s = tid; s < SRC_LEN; s += 256) mx = fmaxf(mx, row[s]);
    red[tid] = mx; __syncthreads();
    for (int off = 128; off > 0; off >>= 1) {
        if (tid < off) red[tid] = fmaxf(red[tid], red[tid + off]);
        __syncthreads();
    }
    mx = red[0]; __syncthreads();

    float sum = 0.0f;
    for (int s = tid; s < SRC_LEN; s += 256) sum += expf(row[s] - mx);
    red[tid] = sum; __syncthreads();
    for (int off = 128; off > 0; off >>= 1) {
        if (tid < off) red[tid] += red[tid + off];
        __syncthreads();
    }
    float inv = 1.0f / red[0];
    for (int s = tid; s < SRC_LEN; s += 256)
        out[(size_t)b * SRC_LEN + s] = expf(row[s] - mx) * inv;
}

// ---------------------------------------------------------------------------
extern "C" void kernel_launch(void* const* d_in, const int* in_sizes, int n_in,
                              void* d_out, int out_size, void* d_ws, size_t ws_size,
                              hipStream_t stream) {
    const float* hidden = (const float*)d_in[0];   // [32,1024]
    const float* enc    = (const float*)d_in[1];   // [2048,32,1024]
    const float* attn_w = (const float*)d_in[2];   // [1024,2048]
    const float* attn_b = (const float*)d_in[3];   // [1024]
    const float* vvec   = (const float*)d_in[4];   // [1024]
    float* out = (float*)d_out;                    // [32,2048]

    char* ws = (char*)d_ws;
    _Float16* wf16  = (_Float16*)ws;                                  // 2 MiB
    float* hpartT   = (float*)(ws + (size_t)DEC_DIM * ENC_DIM * 2);   // 128 KiB
    float* logits   = (float*)(ws + (size_t)DEC_DIM * ENC_DIM * 2
                                  + (size_t)DEC_DIM * BATCH * 4);     // 256 KiB

    wconv_kernel<<<(DEC_DIM * ENC_DIM / 4) / 256, 256, 0, stream>>>(attn_w, wf16);
    hpart_kernel<<<(DEC_DIM * BATCH) / 256, 256, 0, stream>>>(hidden, attn_w, attn_b, hpartT);
    attn_main_kernel<<<ROWS / MBLOCK, 256, SMEM_BYTES, stream>>>(enc, wf16, hpartT, vvec, logits);
    softmax_kernel<<<BATCH, 256, 0, stream>>>(logits, out);
}
